// TokenLinearGL_6141803233516
// MI455X (gfx1250) — matrix operand order repaired and checked by reading, not compiled, not run on hardware
//
#include <hip/hip_runtime.h>
#include <math.h>

typedef __attribute__((ext_vector_type(16))) __bf16 bf16x16;
typedef __attribute__((ext_vector_type(8)))  __bf16 bf16x8;
typedef __attribute__((ext_vector_type(8)))  float  f32x8;

#define TOK_N   1024   // number of tokens T
#define TOK_D   64     // token feature dim
#define IN_F    2048
#define OUT_F   2048
#define ROWS    64     // S-rows per workgroup (simb = 128 KB of 320 KB WGP LDS)
#define NTHREADS 256

__device__ __forceinline__ float gelu_exact(float v) {
    return 0.5f * v * (1.0f + erff(v * 0.70710678118654752f));
}

__device__ __forceinline__ bf16x16 join8(bf16x8 lo, bf16x8 hi) {
    return __builtin_shufflevector(lo, hi, 0,1,2,3,4,5,6,7,8,9,10,11,12,13,14,15);
}

__device__ __forceinline__ bf16x16 cvt16(const float4 a0, const float4 a1,
                                         const float4 a2, const float4 a3) {
    bf16x16 r;
    r[0]=(__bf16)a0.x;  r[1]=(__bf16)a0.y;  r[2]=(__bf16)a0.z;  r[3]=(__bf16)a0.w;
    r[4]=(__bf16)a1.x;  r[5]=(__bf16)a1.y;  r[6]=(__bf16)a1.z;  r[7]=(__bf16)a1.w;
    r[8]=(__bf16)a2.x;  r[9]=(__bf16)a2.y;  r[10]=(__bf16)a2.z; r[11]=(__bf16)a2.w;
    r[12]=(__bf16)a3.x; r[13]=(__bf16)a3.y; r[14]=(__bf16)a3.z; r[15]=(__bf16)a3.w;
    return r;
}

// ---------------------------------------------------------------------------
// Kernel 0: one-shot x (f32) -> xb (bf16). Output stays hot in the 192MB L2.
// ---------------------------------------------------------------------------
__global__ __launch_bounds__(256)
void x_to_bf16_kernel(const float* __restrict__ x, __bf16* __restrict__ xb) {
    const long i = ((long)blockIdx.x * 256 + threadIdx.x) * 8;
    const float4 a = *(const float4*)(x + i);
    const float4 b = *(const float4*)(x + i + 4);
    bf16x8 o;
    o[0]=(__bf16)a.x; o[1]=(__bf16)a.y; o[2]=(__bf16)a.z; o[3]=(__bf16)a.w;
    o[4]=(__bf16)b.x; o[5]=(__bf16)b.y; o[6]=(__bf16)b.z; o[7]=(__bf16)b.w;
    *(bf16x8*)(xb + i) = o;
}

// ---------------------------------------------------------------------------
// Kernel 1: token MLP. One block per token (64 threads).
//   Kt[t][d] = (gelu(tokens @ Wk1) @ Wk2)[t,d]            bf16, row-major
//   Vt[f][t] = (gelu(tokens @ Wv1) @ Wv2)[t,f] transposed bf16 (T-contiguous)
// ---------------------------------------------------------------------------
__global__ __launch_bounds__(64)
void token_mlp_kernel(const float* __restrict__ tokens,
                      const float* __restrict__ Wk1, const float* __restrict__ Wk2,
                      const float* __restrict__ Wv1, const float* __restrict__ Wv2,
                      __bf16* __restrict__ Kt, __bf16* __restrict__ Vt) {
    __shared__ float tok[TOK_D];
    __shared__ float hk[TOK_D];
    __shared__ float hv[TOK_D];
    const int t = blockIdx.x;
    const int j = threadIdx.x;
    tok[j] = tokens[t * TOK_D + j];
    __syncthreads();
    float sk = 0.f, sv = 0.f;
    #pragma unroll 8
    for (int i = 0; i < TOK_D; ++i) {
        const float ti = tok[i];
        sk += ti * Wk1[i * TOK_D + j];
        sv += ti * Wv1[i * TOK_D + j];
    }
    hk[j] = gelu_exact(sk);
    hv[j] = gelu_exact(sv);
    __syncthreads();
    for (int d = j; d < IN_F; d += TOK_D) {
        float a = 0.f, b = 0.f;
        #pragma unroll 8
        for (int i = 0; i < TOK_D; ++i) {
            a += hk[i] * Wk2[i * IN_F + d];
            b += hv[i] * Wv2[i * OUT_F + d];
        }
        Kt[(long)t * IN_F + d]  = (__bf16)a;
        Vt[(long)d * TOK_N + t] = (__bf16)b;    // transposed for WMMA B-frags
    }
}

// ---------------------------------------------------------------------------
// Kernel 2: fused  sim = x@K^T  ->  row scale + gelu  ->  out = W@V
// 256 threads = 8 waves (wave32). Each WG handles ROWS=64 rows of x.
// XBF=true : A-fragments are direct bf16 b128 loads from pre-converted xb.
// XBF=false: A-fragments converted f32->bf16 in-loop (no large ws needed).
// ---------------------------------------------------------------------------
template <bool XBF>
__global__ __launch_bounds__(NTHREADS)
void token_linear_main_kernel(const float*  __restrict__ x,
                              const __bf16* __restrict__ xb,
                              const __bf16* __restrict__ Kt,   // [TOK_N][IN_F]
                              const __bf16* __restrict__ Vt,   // [OUT_F][TOK_N]
                              float* __restrict__ out) {
    __shared__ __bf16 simb[ROWS][TOK_N];   // 128 KB: sim tile, later gelu'd W
    __shared__ float  rowSumSq[ROWS];
    __shared__ float  rowScale[ROWS];

    const int  tid   = threadIdx.x;
    const int  wave  = tid >> 5;
    const int  lane  = tid & 31;
    const int  lhalf = lane >> 4;        // which 16-lane half
    const int  lmod  = lane & 15;
    const long rowBase = (long)blockIdx.x * ROWS;

    if (tid < ROWS) rowSumSq[tid] = 0.f;
    __syncthreads();

    // wave -> (m-tile 0..3, t-half 0..1)
    const int mt  = wave >> 1;
    const int m0  = mt * 16;
    const int t0w = (wave & 1) * 512;    // 512 sim columns per wave

    const long  xoff  = (rowBase + m0 + lmod) * (long)IN_F;
    const float*  xrow  = x  + xoff;
    const __bf16* xbrow = xb + xoff;

    float ss[8];
    #pragma unroll
    for (int r = 0; r < 8; ++r) ss[r] = 0.f;

    // ---------------- Phase A: sim tile = X @ K^T (bf16 WMMA, f32 acc) -----
    for (int g = 0; g < 4; ++g) {                 // 4 groups of 8 t-tiles
        const int tg = t0w + g * 128;
        f32x8 acc[8];
        #pragma unroll
        for (int j = 0; j < 8; ++j)
            acc[j] = (f32x8){0.f,0.f,0.f,0.f,0.f,0.f,0.f,0.f};

        for (int k0 = 0; k0 < IN_F; k0 += 32) {
            // A fragment: row m0+lmod, K chunks [k0+lhalf*8, +8) and +16
            bf16x16 afrag;
            if (XBF) {
                const __bf16* ap = xbrow + k0 + lhalf * 8;
                afrag = join8(*(const bf16x8*)ap, *(const bf16x8*)(ap + 16));
            } else {
                const float* ap = xrow + k0 + lhalf * 8;
                afrag = cvt16(*(const float4*)(ap),      *(const float4*)(ap + 4),
                              *(const float4*)(ap + 16), *(const float4*)(ap + 20));
            }
            #pragma unroll
            for (int j = 0; j < 8; ++j) {
                // B fragment: column t = tile_base + lmod, K chunks [k0+lhalf*8, +8) and +16
                const __bf16* bp = Kt + (long)(tg + j * 16 + lmod) * IN_F
                                      + k0 + lhalf * 8;
                const bf16x16 bfrag = join8(*(const bf16x8*)bp,
                                            *(const bf16x8*)(bp + 16));
                acc[j] = __builtin_amdgcn_wmma_f32_16x16x32_bf16(
                    false, afrag, false, bfrag, (short)0, acc[j], false, false);
            }
        }
        // spill sim tile to LDS (bf16) + accumulate per-row sum of squares
        #pragma unroll
        for (int j = 0; j < 8; ++j) {
            const int col = tg + j * 16 + lmod;
            #pragma unroll
            for (int r = 0; r < 8; ++r) {
                const float v = acc[j][r];
                ss[r] = fmaf(v, v, ss[r]);
                simb[m0 + lhalf * 8 + r][col] = (__bf16)v;
            }
        }
    }
    #pragma unroll
    for (int r = 0; r < 8; ++r)
        atomicAdd(&rowSumSq[m0 + lhalf * 8 + r], ss[r]);   // ds_add_f32
    __syncthreads();

    if (tid < ROWS)
        rowScale[tid] = 32.0f * rsqrtf(rowSumSq[tid]);     // sqrt(T)/||sim||
    __syncthreads();

    // ---------------- Phase B1: W = gelu(sim * scale) in place -------------
    for (int i = tid; i < ROWS * TOK_N; i += NTHREADS) {
        const int rr = i >> 10;
        const int cc = i & (TOK_N - 1);
        const float v = (float)simb[rr][cc] * rowScale[rr];
        simb[rr][cc] = (__bf16)gelu_exact(v);
    }
    __syncthreads();

    // ---------------- Phase B2: out = W @ V (bf16 WMMA, f32 acc) -----------
    const int n0w = (wave & 1) * 1024;            // 1024 out columns per wave
    for (int g = 0; g < 8; ++g) {                 // 8 groups of 8 n-tiles
        const int ng = n0w + g * 128;
        f32x8 acc[8];
        #pragma unroll
        for (int j = 0; j < 8; ++j)
            acc[j] = (f32x8){0.f,0.f,0.f,0.f,0.f,0.f,0.f,0.f};

        for (int k0 = 0; k0 < TOK_N; k0 += 32) {
            // A fragment from LDS weights: row m0+lmod, chunks along T
            const __bf16* ap = &simb[m0 + lmod][k0 + lhalf * 8];
            const bf16x16 afrag = join8(*(const bf16x8*)ap,
                                        *(const bf16x8*)(ap + 16));
            #pragma unroll
            for (int j = 0; j < 8; ++j) {
                // B fragment from V^T: column f = ng + j*16 + lmod, T chunks [k0+lhalf*8, +8) and +16
                const __bf16* bp = Vt + (long)(ng + j * 16 + lmod) * TOK_N
                                      + k0 + lhalf * 8;
                const bf16x16 bfrag = join8(*(const bf16x8*)bp,
                                            *(const bf16x8*)(bp + 16));
                acc[j] = __builtin_amdgcn_wmma_f32_16x16x32_bf16(
                    false, afrag, false, bfrag, (short)0, acc[j], false, false);
            }
        }
        // write f32 output, coalesced across the 16-lane halves
        #pragma unroll
        for (int j = 0; j < 8; ++j) {
            const int col = ng + j * 16 + lmod;
            const long row0 = rowBase + m0 + lhalf * 8;
            #pragma unroll
            for (int r = 0; r < 8; ++r)
                out[(row0 + r) * (long)OUT_F + col] = acc[j][r];
        }
    }
}

// ---------------------------------------------------------------------------
extern "C" void kernel_launch(void* const* d_in, const int* in_sizes, int n_in,
                              void* d_out, int out_size, void* d_ws, size_t ws_size,
                              hipStream_t stream) {
    (void)in_sizes; (void)n_in; (void)out_size;
    const float* x      = (const float*)d_in[0];
    const float* tokens = (const float*)d_in[1];
    const float* Wk1    = (const float*)d_in[2];
    const float* Wk2    = (const float*)d_in[3];
    const float* Wv1    = (const float*)d_in[4];
    const float* Wv2    = (const float*)d_in[5];
    float* out = (float*)d_out;

    const size_t totalRows = 4 * 4096;
    const size_t nKV = (size_t)TOK_N * IN_F + (size_t)OUT_F * TOK_N;   // bf16 elems
    const size_t nX  = totalRows * IN_F;                               // bf16 elems

    // workspace layout: Kt (4MB) | Vt (4MB) | optional xb (64MB)
    __bf16* Kt = (__bf16*)d_ws;
    __bf16* Vt = Kt + (size_t)TOK_N * IN_F;
    __bf16* xb = Kt + nKV;

    token_mlp_kernel<<<TOK_N, 64, 0, stream>>>(tokens, Wk1, Wk2, Wv1, Wv2, Kt, Vt);

    const bool haveXb = ws_size >= (nKV + nX) * sizeof(__bf16);
    if (haveXb) {
        x_to_bf16_kernel<<<(int)(nX / (8 * 256)), 256, 0, stream>>>(x, xb);
        token_linear_main_kernel<true><<<(int)(totalRows / ROWS), NTHREADS, 0, stream>>>(
            x, xb, Kt, Vt, out);
    } else {
        token_linear_main_kernel<false><<<(int)(totalRows / ROWS), NTHREADS, 0, stream>>>(
            x, xb, Kt, Vt, out);
    }
}
